// BipartPool_48284022342135
// MI455X (gfx1250) — matrix operand order repaired and checked by reading, not compiled, not run on hardware
//
#include <hip/hip_runtime.h>
#include <hip/hip_bf16.h>

typedef __attribute__((ext_vector_type(16))) _Float16 v16h;
typedef __attribute__((ext_vector_type(8)))  _Float16 v8h;
typedef __attribute__((ext_vector_type(8)))  float    v8f;
typedef __attribute__((ext_vector_type(2)))  float    v2f;
typedef __attribute__((ext_vector_type(4)))  int      v4i;

#define NUM_GRAPHS 64
#define PTS        512
#define RATIO      16
#define HEADS      4
#define CH         64
#define NTOT       (NUM_GRAPHS * PTS)   // 32768
#define COUT       (HEADS * CH)         // 256
#define NC         (NUM_GRAPHS * RATIO) // 1024
#define NEG_SLOPE  0.2f
#define LROW       72                   // x_l LDS row stride in halves (144 B, 16B aligned)

#define AS1 __attribute__((address_space(1)))
#define AS3 __attribute__((address_space(3)))

#if __has_builtin(__builtin_amdgcn_global_load_async_to_lds_b128)
#define USE_ASYNC_LDS 1
#else
#define USE_ASYNC_LDS 0
#endif

__device__ __forceinline__ void wait_async_zero() {
#if __has_builtin(__builtin_amdgcn_s_wait_asynccnt)
  __builtin_amdgcn_s_wait_asynccnt(0);
#else
  asm volatile("s_wait_asynccnt 0x0" ::: "memory");
#endif
}

// ---------------------------------------------------------------------------
// k0: x_r base = xcent_base(16x64) @ W_r(64x256) + b_r  (tiny; only 16 rows)
// ---------------------------------------------------------------------------
__global__ void k_xr(const float* __restrict__ xcb, const float* __restrict__ Wr,
                     const float* __restrict__ br, float* __restrict__ xr) {
  for (int e = threadIdx.x; e < RATIO * COUT; e += blockDim.x) {
    int r = e >> 8, col = e & 255;
    float acc = br[col];
    for (int k = 0; k < CH; ++k)
      acc += xcb[r * CH + k] * Wr[k * COUT + col];
    xr[e] = acc;
  }
}

// ---------------------------------------------------------------------------
// k1: x_l = x(32768x64) @ W_l(64x256) + b_l, stored f16.
// One wave per 16x16 output tile; K=64 -> 2x v_wmma_f32_16x16x32_f16.
// ---------------------------------------------------------------------------
__global__ __launch_bounds__(128) void k_proj_xl(const float* __restrict__ x,
                                                 const float* __restrict__ Wl,
                                                 const float* __restrict__ bl,
                                                 _Float16* __restrict__ xl) {
  const int tid  = threadIdx.x;
  const int tile = blockIdx.x * 4 + (tid >> 5);
  const int lane = tid & 31;
  const int row0 = (tile >> 4) << 4;   // 2048 row tiles
  const int col0 = (tile & 15) << 4;   // 16 col tiles
  const int m  = lane & 15;
  const int hi = lane >> 4;
  const int n  = col0 + m;

  v8f acc = {};
#pragma unroll
  for (int kstep = 0; kstep < 2; ++kstep) {
    v16h a, bf;
#pragma unroll
    for (int p = 0; p < 8; ++p) {          // A 16x32 f16 layout
      int K = kstep * 32 + ((p < 4) ? 0 : 16) + hi * 8 + 2 * (p & 3);
      v2f v = *(const v2f*)(x + (row0 + m) * CH + K);
      a[2 * p]     = (_Float16)v.x;
      a[2 * p + 1] = (_Float16)v.y;
    }
#pragma unroll
    for (int q = 0; q < 8; ++q) {          // B 32x16 f16 layout
      int k = kstep * 32 + ((q < 4) ? 0 : 16) + hi * 8 + 2 * (q & 3);
      bf[2 * q]     = (_Float16)Wl[k * COUT + n];
      bf[2 * q + 1] = (_Float16)Wl[(k + 1) * COUT + n];
    }
    acc = __builtin_amdgcn_wmma_f32_16x16x32_f16(false, a, false, bf,
                                                 (short)0, acc, false, false);
  }
  const float bv = bl[n];
#pragma unroll
  for (int i = 0; i < 8; ++i) {            // C layout: rows i (lo) / i+8 (hi)
    int row = row0 + i + 8 * hi;
    xl[row * COUT + n] = (_Float16)(acc[i] + bv);
  }
}

// ---------------------------------------------------------------------------
// k2: per (graph b, head h): GATv2 attention + softmax + WMMA aggregation.
// LDS: x_l slice row-major padded [s][LROW] (72 KB f16, async-staged) +
//      alpha/w [r][s] (32 KB f32).
// ---------------------------------------------------------------------------
__global__ __launch_bounds__(128) void k_attn_agg(const _Float16* __restrict__ xl,
                                                  const float* __restrict__ xrb,
                                                  const float* __restrict__ att,
                                                  float* __restrict__ outh) {
  const int b   = blockIdx.x >> 2;
  const int h   = blockIdx.x & 3;
  const int tid = threadIdx.x;

  __shared__ _Float16 xlS[PTS * LROW];    // [s][c], 144B row stride -> 72 KB
  __shared__ float alpha[RATIO * PTS];    // [r][s]  32 KB (alpha -> e -> w)
  __shared__ float xrS[RATIO * CH];
  __shared__ float attS[CH];
  __shared__ float red[RATIO * 8];
  __shared__ float mS[RATIO], dInv[RATIO], wSelf[RATIO], aSelf[RATIO], eSelf[RATIO];

  const int s0 = b * PTS;

  // Stage x_l[b-slice, head h] into LDS (async DMA path when available).
#if USE_ASYNC_LDS
  for (int idx = tid; idx < PTS * 8; idx += 128) {
    int s = idx >> 3, c8 = idx & 7;
    __builtin_amdgcn_global_load_async_to_lds_b128(
        (AS1 v4i*)(xl + (size_t)(s0 + s) * COUT + h * CH + c8 * 8),
        (AS3 v4i*)&xlS[s * LROW + c8 * 8], 0, 0);
  }
  wait_async_zero();
#else
  for (int idx = tid; idx < PTS * 8; idx += 128) {
    int s = idx >> 3, c8 = idx & 7;
    *(v8h*)&xlS[s * LROW + c8 * 8] =
        *(const v8h*)(xl + (size_t)(s0 + s) * COUT + h * CH + c8 * 8);
  }
#endif
  for (int idx = tid; idx < RATIO * CH; idx += 128)
    xrS[idx] = xrb[(idx >> 6) * COUT + h * CH + (idx & 63)];
  if (tid < CH) attS[tid] = att[h * CH + tid];
  __syncthreads();

  // Phase 1: alpha[r][s] = att . leaky_relu(xl[s] + xr[r])
  for (int ii = 0; ii < 4; ++ii) {
    int s = tid + 128 * ii;
    float acc[RATIO];
#pragma unroll
    for (int r = 0; r < RATIO; ++r) acc[r] = 0.f;
    for (int c = 0; c < CH; ++c) {
      float xv = (float)xlS[s * LROW + c];
      float av = attS[c];
#pragma unroll
      for (int r = 0; r < RATIO; ++r) {
        float t = xv + xrS[r * CH + c];
        t = (t > 0.f) ? t : NEG_SLOPE * t;
        acc[r] += t * av;
      }
    }
#pragma unroll
    for (int r = 0; r < RATIO; ++r) {
      float a = acc[r];
      if (b == 0 && s == r) a = -1e30f;    // PyG drops src==dst edges
      alpha[r * PTS + s] = a;
    }
  }
  // Self-loop edge: source = global node t = b*16 + r (cross-graph for t>=16)
  if (tid < RATIO) {
    int g = b * RATIO + tid;
    float acc = 0.f;
    for (int c = 0; c < CH; ++c) {
      float t = (float)xl[(size_t)g * COUT + h * CH + c] + xrS[tid * CH + c];
      t = (t > 0.f) ? t : NEG_SLOPE * t;
      acc += t * attS[c];
    }
    aSelf[tid] = acc;
  }
  __syncthreads();

  // Softmax per centroid row r: max
  {
    int r = tid >> 3, j = tid & 7;
    float pm = -3.4e38f;
    for (int k = 0; k < PTS / 8; ++k)
      pm = fmaxf(pm, alpha[r * PTS + j + 8 * k]);
    red[r * 8 + j] = pm;
  }
  __syncthreads();
  if (tid < RATIO) {
    float m = aSelf[tid];
    for (int j = 0; j < 8; ++j) m = fmaxf(m, red[tid * 8 + j]);
    mS[tid] = m;
    eSelf[tid] = __expf(aSelf[tid] - m);
  }
  __syncthreads();
  // exp + sum
  {
    int r = tid >> 3, j = tid & 7;
    float ps = 0.f, m = mS[r];
    for (int k = 0; k < PTS / 8; ++k) {
      int idx = r * PTS + j + 8 * k;
      float e = __expf(alpha[idx] - m);
      alpha[idx] = e;
      ps += e;
    }
    red[r * 8 + j] = ps;
  }
  __syncthreads();
  if (tid < RATIO) {
    float d = eSelf[tid];
    for (int j = 0; j < 8; ++j) d += red[tid * 8 + j];
    float inv = 1.f / d;
    dInv[tid]  = inv;
    wSelf[tid] = eSelf[tid] * inv;
  }
  __syncthreads();
  for (int t2 = tid; t2 < RATIO * PTS; t2 += 128)
    alpha[t2] *= dInv[t2 >> 9];
  __syncthreads();

  // Phase 3: out(16x64) = W(16x512) @ XL(512x64); one 16-col tile per wave.
  const int ctile = tid >> 5;
  const int lane  = tid & 31;
  const int nn = lane & 15, hi = lane >> 4;
  v8f acc = {};
#pragma unroll
  for (int kstep = 0; kstep < PTS / 32; ++kstep) {
    v16h a, bf;
#pragma unroll
    for (int p = 0; p < 8; ++p) {
      int kk = kstep * 32 + ((p < 4) ? 0 : 16) + hi * 8 + 2 * (p & 3);
      a[2 * p]     = (_Float16)alpha[nn * PTS + kk];
      a[2 * p + 1] = (_Float16)alpha[nn * PTS + kk + 1];
    }
#pragma unroll
    for (int q = 0; q < 8; ++q) {
      int kk = kstep * 32 + ((q < 4) ? 0 : 16) + hi * 8 + 2 * (q & 3);
      int c  = ctile * 16 + nn;
      bf[2 * q]     = xlS[kk * LROW + c];
      bf[2 * q + 1] = xlS[(kk + 1) * LROW + c];
    }
    acc = __builtin_amdgcn_wmma_f32_16x16x32_f16(false, a, false, bf,
                                                 (short)0, acc, false, false);
  }
  // Epilogue: add self-loop contribution, write per-head output.
#pragma unroll
  for (int i = 0; i < 8; ++i) {
    int m = i + 8 * hi;
    int g = b * RATIO + m;
    float val = acc[i] +
        wSelf[m] * (float)xl[(size_t)g * COUT + h * CH + ctile * 16 + nn];
    outh[((b * HEADS + h) * RATIO + m) * CH + ctile * 16 + nn] = val;
  }
}

// ---------------------------------------------------------------------------
// k3: mean over heads + bias; append int64 batchcent after the f32 block.
// ---------------------------------------------------------------------------
__global__ void k_final(const float* __restrict__ outh, const float* __restrict__ bias,
                        float* __restrict__ out) {
  int idx = blockIdx.x * blockDim.x + threadIdx.x;
  if (idx < NC * CH) {
    int t = idx >> 6, c = idx & 63;
    int b = t >> 4,  r = t & 15;
    float s = 0.f;
    for (int h = 0; h < HEADS; ++h)
      s += outh[((b * HEADS + h) * RATIO + r) * CH + c];
    out[idx] = 0.25f * s + bias[c];
  }
  if (idx < NC) {
    long long* bc = (long long*)(out + NC * CH);
    bc[idx] = (long long)(idx >> 4);
  }
}

extern "C" void kernel_launch(void* const* d_in, const int* in_sizes, int n_in,
                              void* d_out, int out_size, void* d_ws, size_t ws_size,
                              hipStream_t stream) {
  (void)in_sizes; (void)n_in; (void)out_size; (void)ws_size;
  const float* x    = (const float*)d_in[0];
  const float* xcb  = (const float*)d_in[3];
  const float* Wl   = (const float*)d_in[4];
  const float* bl   = (const float*)d_in[5];
  const float* Wr   = (const float*)d_in[6];
  const float* br   = (const float*)d_in[7];
  const float* att  = (const float*)d_in[8];
  const float* bias = (const float*)d_in[9];

  char* ws = (char*)d_ws;
  _Float16* xl = (_Float16*)ws;                                   // 16 MB
  float* xrb   = (float*)(ws + (size_t)NTOT * COUT * 2);          // 16 KB
  float* outh  = (float*)(ws + (size_t)NTOT * COUT * 2 + RATIO * COUT * 4); // 1 MB

  k_xr<<<1, 256, 0, stream>>>(xcb, Wr, br, xrb);
  k_proj_xl<<<(NTOT / 16) * (COUT / 16) / 4, 128, 0, stream>>>(x, Wl, bl, xl);
  k_attn_agg<<<NUM_GRAPHS * HEADS, 128, 0, stream>>>(xl, xrb, att, outh);
  k_final<<<(NC * CH + 255) / 256, 256, 0, stream>>>(outh, bias, (float*)d_out);
}
